// GEAR_43731357008172
// MI455X (gfx1250) — compile-verified, matrix operand order checked by reading, not run on hardware
//
#include <hip/hip_runtime.h>

// ---------------------------------------------------------------------------
// Problem constants (from reference): B=8192, NINS=5, NF=768, NLAYER=2,
// NCLASS=3, NH=64.  Output: (B,3) fp32 log-softmax.
// ---------------------------------------------------------------------------
#define NB      8192
#define NINS    5
#define NF      768
#define NH      64
#define NCLASS  3
#define BT      16          // batch rows per workgroup
#define BDIM    256         // 8 waves (wave32)
#define KSTEPS  (NF / 32)   // 24 wmma K-steps of 32
#define MATELEMS (NF * NH)  // 49152 elements per 768x64 weight matrix
#define NMATS   22          // 2 layers * 5 q * {W1a,W1b} + aggW1a + aggW1b

typedef __bf16 bf16_t;
typedef bf16_t bfv16 __attribute__((ext_vector_type(16)));
typedef float  fv8   __attribute__((ext_vector_type(8)));
typedef float  fv4   __attribute__((ext_vector_type(4)));
typedef unsigned int uv4 __attribute__((ext_vector_type(4)));
typedef unsigned int uv2 __attribute__((ext_vector_type(2)));
typedef unsigned short bfr;   // raw bf16 storage

union FragBF { uv4 u[2]; bfv16 v; };

__device__ __forceinline__ unsigned int f2bf(float f) {
    unsigned int u = __float_as_uint(f);
    return ((u + 0x7FFFu + ((u >> 16) & 1u)) >> 16) & 0xFFFFu;   // RNE
}
__device__ __forceinline__ float bf2f(unsigned int h) {
    return __uint_as_float(h << 16);
}

// ---------------------------------------------------------------------------
// Prep kernel: convert W1 / aggW1 fp32 -> bf16, pre-swizzled into WMMA
// B-fragment order.  Fragment (kk,nt): lane holds column n = nt*16+(lane&15),
// its 16 K-values (pattern e + (e>=8?8:0) + (lane>=16?8:0)) stored contiguous
// so the GEMM loads each lane's fragment with two 16B loads.
// ---------------------------------------------------------------------------
__global__ void GEAR_convert_weights(const float* __restrict__ W1,
                                     const float* __restrict__ aggW1,
                                     bfr* __restrict__ out) {
    int idx = blockIdx.x * BDIM + threadIdx.x;
    if (idx >= NMATS * MATELEMS) return;
    int matid = idx / MATELEMS;
    int r     = idx % MATELEMS;
    int e     = r & 15;
    int lane  = (r >> 4) & 31;
    int fn    = r >> 9;            // kk*4 + nt
    int nt    = fn & 3;
    int kk    = fn >> 2;
    int k = kk * 32 + e + ((e & 8) ? 8 : 0) + ((lane & 16) ? 8 : 0);
    int n = nt * 16 + (lane & 15);
    float src;
    if (matid < 20) {
        int l   = matid / 10;
        int q   = (matid % 10) >> 1;
        int isB = matid & 1;                 // 0 = W1a (own), 1 = W1b (key)
        src = W1[(((l * NINS + q) * (2 * NF)) + isB * NF + k) * NH + n];
    } else {
        src = aggW1[((matid - 20) * NF + k) * NH + n];
    }
    out[idx] = (bfr)f2bf(src);
}

// ---------------------------------------------------------------------------
// 24-tile WMMA GEMM block: tiles 0..19 -> P = A80(80x768) * Wp(768x64),
// tiles 20..23 -> O = Ao(16x768) * Wo(768x64).  Results in Pb[96][64] fp32
// (P rows 0..79 = (bt*5+k), O rows 80..95 = bt).
// ---------------------------------------------------------------------------
__device__ __forceinline__ void gemm24(const bfr* __restrict__ pA,   // 80 rows, stride NF
                                       const bfr* __restrict__ oA,   // 16 rows
                                       int oAstride,
                                       const bfr* __restrict__ wP,   // swizzled key weights
                                       const bfr* __restrict__ wO,   // swizzled own weights
                                       float* __restrict__ Pb, int tid) {
    const int lane  = tid & 31;
    const int wid   = tid >> 5;
    const int hioff = (lane & 16) ? 8 : 0;
    const int mrow  = lane & 15;
    for (int tile = wid; tile < 24; tile += 8) {
        const bool isO = tile >= 20;
        const int  nt  = isO ? (tile - 20) : (tile & 3);
        const bfr* Ab  = isO ? (oA + mrow * oAstride)
                             : (pA + ((tile >> 2) * 16 + mrow) * NF);
        const bfr* Wb  = isO ? wO : wP;
        fv8 acc = {};
        for (int kk = 0; kk < KSTEPS; ++kk) {
            FragBF a, b;
            a.u[0] = *(const uv4*)(Ab + kk * 32 + hioff);        // K = hioff..hioff+7
            a.u[1] = *(const uv4*)(Ab + kk * 32 + hioff + 16);   // K = 16+hioff..
            const bfr* bp = Wb + ((kk * 4 + nt) * 32 + lane) * 16;
            b.u[0] = *(const uv4*)(bp);
            b.u[1] = *(const uv4*)(bp + 8);
            acc = __builtin_amdgcn_wmma_f32_16x16x32_bf16(
                      false, a.v, false, b.v, (short)0, acc, false, false);
        }
        const int prow0 = isO ? 80 : (tile >> 2) * 16;
        const int M     = (lane & 16) ? 8 : 0;
        const int N     = nt * 16 + mrow;
#pragma unroll
        for (int r = 0; r < 8; ++r)
            Pb[(prow0 + M + r) * NH + N] = acc[r];
    }
}

// ---------------------------------------------------------------------------
// Fully fused forward: 2 attention layers + aggregate + head, one WG = 16 rows.
// ---------------------------------------------------------------------------
__global__ __launch_bounds__(BDIM, 1)
void GEAR_fused(const float* __restrict__ inputs, const float* __restrict__ claims,
                const float* __restrict__ b1, const float* __restrict__ W2,
                const float* __restrict__ b2,
                const float* __restrict__ aggb1, const float* __restrict__ aggW2,
                const float* __restrict__ aggb2,
                const float* __restrict__ Wout, const float* __restrict__ bout,
                const bfr* __restrict__ wsw, float* __restrict__ out) {
    __shared__ __align__(16) bfr   xs[2][BT * NINS * NF];   // 2 x 120 KB bf16 x
    __shared__ __align__(16) float Pb[96 * NH];             // proj results
    __shared__ float sbuf[96];
    __shared__ float wq[NINS][BT][NINS];

    const int tid = threadIdx.x;
    const int b0  = blockIdx.x * BT;

    // ---- load + convert x tile: (16,5,768) fp32 -> bf16 LDS ----
    {
        const fv4* xin = (const fv4*)(inputs + (size_t)b0 * NINS * NF);
        for (int i = tid; i < BT * NINS * NF / 4; i += BDIM) {
            fv4 v = xin[i];
            uv2 p;
            p.x = f2bf(v.x) | (f2bf(v.y) << 16);
            p.y = f2bf(v.z) | (f2bf(v.w) << 16);
            ((uv2*)xs[0])[i] = p;
        }
    }
    __syncthreads();

    int cur = 0;
    // ================= attention layers =================
    for (int l = 0; l < 2; ++l) {
        for (int q = 0; q < NINS; ++q) {
            const int m = l * NINS + q;
            gemm24(xs[cur], &xs[cur][q * NF], NINS * NF,
                   wsw + (size_t)(m * 2 + 1) * MATELEMS,   // W1b (key)
                   wsw + (size_t)(m * 2 + 0) * MATELEMS,   // W1a (own)
                   Pb, tid);
            __syncthreads();
            if (tid < BT * NINS) {                         // score(bt,k)
                const int bt = tid / NINS, k = tid % NINS;
                const float* b1v = b1 + m * NH;
                const float* w2v = W2 + m * NH;
                const float* Or  = Pb + (80 + bt) * NH;
                const float* Pr  = Pb + (bt * NINS + k) * NH;
                float s = b2[m];
                for (int h = 0; h < NH; ++h) {
                    float hv = Or[h] + Pr[h] + b1v[h];
                    hv = hv > 0.f ? hv : 0.f;
                    s += hv * w2v[h];
                }
                sbuf[tid] = s;
            }
            __syncthreads();
            if (tid < BT) {                                // softmax over k
                float mx = -1e30f, e[NINS], sum = 0.f;
                for (int k = 0; k < NINS; ++k) mx = fmaxf(mx, sbuf[tid * NINS + k]);
                for (int k = 0; k < NINS; ++k) { e[k] = __expf(sbuf[tid * NINS + k] - mx); sum += e[k]; }
                for (int k = 0; k < NINS; ++k) wq[q][tid][k] = e[k] / sum;
            }
            __syncthreads();
        }
        // ---- mix: newx[bt,q,:] = sum_k w * x[bt,k,:] -> other buffer ----
        {
            bfr* xc = xs[cur];
            bfr* xn = xs[cur ^ 1];
            for (int idx = tid; idx < BT * NINS * (NF / 8); idx += BDIM) {
                const int bt  = idx / (NINS * (NF / 8));
                const int rem = idx - bt * NINS * (NF / 8);
                const int qq  = rem / (NF / 8);
                const int fc  = rem - qq * (NF / 8);
                const bfr* xr = xc + bt * NINS * NF + fc * 8;
                float ac[8] = {0, 0, 0, 0, 0, 0, 0, 0};
#pragma unroll
                for (int k = 0; k < NINS; ++k) {
                    const float wv = wq[qq][bt][k];
                    uv4 xv = *(const uv4*)(xr + k * NF);
                    ac[0] += wv * bf2f(xv.x & 0xFFFFu); ac[1] += wv * bf2f(xv.x >> 16);
                    ac[2] += wv * bf2f(xv.y & 0xFFFFu); ac[3] += wv * bf2f(xv.y >> 16);
                    ac[4] += wv * bf2f(xv.z & 0xFFFFu); ac[5] += wv * bf2f(xv.z >> 16);
                    ac[6] += wv * bf2f(xv.w & 0xFFFFu); ac[7] += wv * bf2f(xv.w >> 16);
                }
                uv4 o;
                o.x = f2bf(ac[0]) | (f2bf(ac[1]) << 16);
                o.y = f2bf(ac[2]) | (f2bf(ac[3]) << 16);
                o.z = f2bf(ac[4]) | (f2bf(ac[5]) << 16);
                o.w = f2bf(ac[6]) | (f2bf(ac[7]) << 16);
                *(uv4*)(xn + (bt * NINS + qq) * NF + fc * 8) = o;
            }
        }
        __syncthreads();
        cur ^= 1;
    }
    // ================= aggregate =================
    bfr* cb = xs[cur ^ 1];                     // claims tile bf16 (buffer is free)
    {
        const fv4* cin = (const fv4*)(claims + (size_t)b0 * NF);
        for (int i = tid; i < BT * NF / 4; i += BDIM) {
            fv4 v = cin[i];
            uv2 p;
            p.x = f2bf(v.x) | (f2bf(v.y) << 16);
            p.y = f2bf(v.z) | (f2bf(v.w) << 16);
            ((uv2*)cb)[i] = p;
        }
    }
    __syncthreads();
    gemm24(xs[cur], cb, NF,
           wsw + (size_t)21 * MATELEMS,        // aggW1b (keys from x)
           wsw + (size_t)20 * MATELEMS,        // aggW1a (query from claims)
           Pb, tid);
    __syncthreads();
    if (tid < BT * NINS) {
        const int bt = tid / NINS, k = tid % NINS;
        const float* Or = Pb + (80 + bt) * NH;
        const float* Pr = Pb + (bt * NINS + k) * NH;
        float s = aggb2[0];
        for (int h = 0; h < NH; ++h) {
            float hv = Or[h] + Pr[h] + aggb1[h];
            hv = hv > 0.f ? hv : 0.f;
            s += hv * aggW2[h];
        }
        sbuf[tid] = s;
    }
    __syncthreads();
    if (tid < BT) {
        float mx = -1e30f, e[NINS], sum = 0.f;
        for (int k = 0; k < NINS; ++k) mx = fmaxf(mx, sbuf[tid * NINS + k]);
        for (int k = 0; k < NINS; ++k) { e[k] = __expf(sbuf[tid * NINS + k] - mx); sum += e[k]; }
        for (int k = 0; k < NINS; ++k) wq[0][tid][k] = e[k] / sum;
    }
    __syncthreads();
    // pooled[bt,f] = sum_k w * x[bt,k,f]  (fp32, reuse claims buffer)
    float* pooled = (float*)cb;
    for (int idx = tid; idx < BT * NF; idx += BDIM) {
        const int bt = idx / NF, f = idx % NF;
        float s = 0.f;
#pragma unroll
        for (int k = 0; k < NINS; ++k)
            s += wq[0][bt][k] * bf2f((unsigned int)xs[cur][(bt * NINS + k) * NF + f]);
        pooled[idx] = s;
    }
    __syncthreads();
    // ---- head: relu(pooled @ Wout + bout) ----
    if (tid < BT * NCLASS) {
        const int bt = tid / NCLASS, c = tid % NCLASS;
        float s = bout[c];
        const float* pr = pooled + bt * NF;
        for (int f = 0; f < NF; ++f) s += pr[f] * Wout[f * NCLASS + c];
        sbuf[tid] = s > 0.f ? s : 0.f;
    }
    __syncthreads();
    if (tid < BT) {
        float l0 = sbuf[tid * 3 + 0], l1 = sbuf[tid * 3 + 1], l2 = sbuf[tid * 3 + 2];
        float mx  = fmaxf(l0, fmaxf(l1, l2));
        float lse = mx + __logf(__expf(l0 - mx) + __expf(l1 - mx) + __expf(l2 - mx));
        out[(size_t)(b0 + tid) * 3 + 0] = l0 - lse;
        out[(size_t)(b0 + tid) * 3 + 1] = l1 - lse;
        out[(size_t)(b0 + tid) * 3 + 2] = l2 - lse;
    }
}

extern "C" void kernel_launch(void* const* d_in, const int* in_sizes, int n_in,
                              void* d_out, int out_size, void* d_ws, size_t ws_size,
                              hipStream_t stream) {
    (void)in_sizes; (void)n_in; (void)out_size; (void)ws_size;
    const float* inputs = (const float*)d_in[0];
    const float* claims = (const float*)d_in[1];
    const float* W1     = (const float*)d_in[2];
    const float* b1     = (const float*)d_in[3];
    const float* W2     = (const float*)d_in[4];
    const float* b2     = (const float*)d_in[5];
    const float* aggW1  = (const float*)d_in[6];
    const float* aggb1  = (const float*)d_in[7];
    const float* aggW2  = (const float*)d_in[8];
    const float* aggb2  = (const float*)d_in[9];
    const float* Wout   = (const float*)d_in[10];
    const float* bout   = (const float*)d_in[11];
    bfr*   wsw = (bfr*)d_ws;          // 22 * 49152 bf16 = ~2.1 MB scratch
    float* out = (float*)d_out;

    const int tot = NMATS * MATELEMS;
    GEAR_convert_weights<<<(tot + BDIM - 1) / BDIM, BDIM, 0, stream>>>(W1, aggW1, wsw);
    GEAR_fused<<<NB / BT, BDIM, 0, stream>>>(inputs, claims, b1, W2, b2,
                                             aggb1, aggW2, aggb2, Wout, bout,
                                             wsw, out);
}
